// ContrastiveLoss_76501957477132
// MI455X (gfx1250) — matrix operand order repaired and checked by reading, not compiled, not run on hardware
//
#include <hip/hip_runtime.h>

typedef __attribute__((ext_vector_type(16))) __bf16 v16bf;
typedef __attribute__((ext_vector_type(8)))  __bf16 v8bf;
typedef __attribute__((ext_vector_type(8)))  float  v8f;

#define CL_M        8192
#define CL_NTRK     512
#define CL_Q        8
#define CL_D        128
#define CL_NQ       (CL_NTRK * CL_Q)      /* 4096 */
/* exp(s/T) = exp2(s * log2(e)/T), T = 0.3 */
#define CL_EXP2_SCALE  4.8089834696296783f

#define ROW_TILES       (CL_M / 16)       /* 512 */
#define XY_COL_TILES    (CL_NQ / 16)      /* 256 */
#define XX_COL_TILES    (CL_M / 16)       /* 512 */
#define CHUNKS          8
#define XY_TPC          (XY_COL_TILES / CHUNKS)  /* 32 tiles per chunk */
#define XX_TPC          (XX_COL_TILES / CHUNKS)  /* 64 tiles per chunk */

/* ---- workspace layout ----
   float partials (each slot written by exactly one wave -> deterministic):   */
#define WS_XY_TOT   0
#define WS_XY_POS   (WS_XY_TOT  + CL_M * CHUNKS)
#define WS_XX_TOT   (WS_XY_POS  + CL_M * CHUNKS)
#define WS_XX_SAME  (WS_XX_TOT  + CL_M * CHUNKS)
#define WS_XX_DIAG  (WS_XX_SAME + CL_M * CHUNKS)
#define WS_PART_FLOATS (WS_XX_DIAG + CL_M * CHUNKS)          /* 327680 floats */
/* bf16 hi/lo mirrors of x and yf, appended after the partials (byte offsets) */
#define WS_XHI_OFF  ((size_t)WS_PART_FLOATS * 4)             /* 1,310,720 B   */
#define WS_XLO_OFF  (WS_XHI_OFF + (size_t)CL_M  * CL_D * 2)  /* +2 MB         */
#define WS_YHI_OFF  (WS_XLO_OFF + (size_t)CL_M  * CL_D * 2)  /* +2 MB         */
#define WS_YLO_OFF  (WS_YHI_OFF + (size_t)CL_NQ * CL_D * 2)  /* +1 MB         */
#define WS_BYTES    (WS_YLO_OFF + (size_t)CL_NQ * CL_D * 2)  /* ~7.25 MB      */

/* ---------------- fp32 -> bf16 hi/lo split (done ONCE, outside the GEMMs) -- */
__global__ __launch_bounds__(256)
void cl_convert_kernel(const float* __restrict__ src,
                       unsigned long long* __restrict__ hi64,
                       unsigned long long* __restrict__ lo64, int n4) {
    const int i = blockIdx.x * blockDim.x + threadIdx.x;
    if (i >= n4) return;
    const float4 v = ((const float4*)src)[i];
    unsigned long long hp = 0ull, lp = 0ull;
    const float f[4] = {v.x, v.y, v.z, v.w};
#pragma unroll
    for (int k = 0; k < 4; ++k) {
        unsigned u = __builtin_bit_cast(unsigned, f[k]);
        unsigned hb = u >> 16;                               /* truncated bf16 */
        float hf = __builtin_bit_cast(float, hb << 16);
        unsigned lb = __builtin_bit_cast(unsigned, f[k] - hf) >> 16;
        hp |= ((unsigned long long)(hb & 0xffffu)) << (16 * k);
        lp |= ((unsigned long long)(lb & 0xffffu)) << (16 * k);
    }
    hi64[i] = hp;   /* 4 bf16 packed -> single b64 store */
    lo64[i] = lp;
}

static __device__ inline v16bf cat8(v8bf a, v8bf b) {
    return __builtin_shufflevector(a, b, 0, 1, 2, 3, 4, 5, 6, 7,
                                         8, 9, 10, 11, 12, 13, 14, 15);
}

/* A fragment (16x32 bf16): lane (r,h) row = r; elems 0..7 = K 32t+8h+0..7,
   elems 8..15 = K 32t+16+8h+0..7. Pure b128 loads from pre-split bf16. */
static __device__ inline v16bf load_frag_A_bf(const __bf16* __restrict__ rowPtr,
                                              int t, int h) {
    const v8bf a0 = *(const v8bf*)(rowPtr + t * 32 + h * 8);
    const v8bf a1 = *(const v8bf*)(rowPtr + t * 32 + h * 8 + 16);
    return cat8(a0, a1);
}

/* B fragment (32x16 bf16): lane (r,h) "column" row = r; elems 0..7 = K 32t+8h+0..7,
   elems 8..15 = K 32t+16+8h+0..7 */
static __device__ inline v16bf load_frag_B_bf(const __bf16* __restrict__ rowPtr,
                                              int t, int h) {
    const v8bf b0 = *(const v8bf*)(rowPtr + t * 32 + h * 8);
    const v8bf b1 = *(const v8bf*)(rowPtr + t * 32 + h * 8 + 16);
    return cat8(b0, b1);
}

/* one 16x16 tile: 4 K-chunks x (hi*hi + hi*lo + lo*hi) = 12 WMMAs */
static __device__ inline v8f tile_mm(const __bf16* __restrict__ browh,
                                     const __bf16* __restrict__ browl,
                                     const v16bf* __restrict__ Ahi,
                                     const v16bf* __restrict__ Alo, int h) {
    v8f c = {0.0f, 0.0f, 0.0f, 0.0f, 0.0f, 0.0f, 0.0f, 0.0f};
#pragma unroll
    for (int t = 0; t < 4; ++t) {
        const v16bf Bhi = load_frag_B_bf(browh, t, h);
        const v16bf Blo = load_frag_B_bf(browl, t, h);
        c = __builtin_amdgcn_wmma_f32_16x16x32_bf16(false, Ahi[t], false, Bhi, (short)0, c, false, false);
        c = __builtin_amdgcn_wmma_f32_16x16x32_bf16(false, Ahi[t], false, Blo, (short)0, c, false, false);
        c = __builtin_amdgcn_wmma_f32_16x16x32_bf16(false, Alo[t], false, Bhi, (short)0, c, false, false);
    }
    return c;
}

/* ---------------- x @ yf^T term ---------------- */
__global__ __launch_bounds__(256)
void cl_xy_kernel(const __bf16* __restrict__ xhi, const __bf16* __restrict__ xlo,
                  const __bf16* __restrict__ yhi, const __bf16* __restrict__ ylo,
                  const int* __restrict__ trkIdx, float* __restrict__ ws) {
    const int lane = threadIdx.x & 31;
    const int wave = blockIdx.x * (blockDim.x >> 5) + (threadIdx.x >> 5);
    const int r = lane & 15;
    const int h = lane >> 4;
    const int rowTile  = wave >> 3;       /* 512 row tiles */
    const int colChunk = wave & 7;        /* 8 chunks of 32 col tiles */
    const int rowBase  = rowTile * 16;

    v16bf Ahi[4], Alo[4];
    const __bf16* arowh = xhi + (size_t)(rowBase + r) * CL_D;
    const __bf16* arowl = xlo + (size_t)(rowBase + r) * CL_D;
#pragma unroll
    for (int t = 0; t < 4; ++t) {
        Ahi[t] = load_frag_A_bf(arowh, t, h);
        Alo[t] = load_frag_A_bf(arowl, t, h);
    }

    int trk[8];
#pragma unroll
    for (int j = 0; j < 8; ++j) trk[j] = trkIdx[rowBase + 8 * h + j];

    float tot[8], pos[8];
#pragma unroll
    for (int j = 0; j < 8; ++j) { tot[j] = 0.0f; pos[j] = 0.0f; }

    const int ctBeg = colChunk * XY_TPC;
    const int ctEnd = ctBeg + XY_TPC;
    /* two independent tiles per iteration: tile1's XDL work fills the
       WMMA->VALU hazard slots of tile0's exp epilogue (and vice versa) */
    for (int ct = ctBeg; ct < ctEnd; ct += 2) {
        const int col0 = ct * 16 + r;
        const int col1 = col0 + 16;
        const __bf16* b0h = yhi + (size_t)col0 * CL_D;
        const __bf16* b0l = ylo + (size_t)col0 * CL_D;
        const __bf16* b1h = yhi + (size_t)col1 * CL_D;
        const __bf16* b1l = ylo + (size_t)col1 * CL_D;
        if (ct + 2 < ctEnd) {
            __builtin_prefetch(b0h + 32 * CL_D, 0, 1);
            __builtin_prefetch(b0l + 32 * CL_D, 0, 1);
        }
        const v8f c0 = tile_mm(b0h, b0l, Ahi, Alo, h);
        const v8f c1 = tile_mm(b1h, b1l, Ahi, Alo, h);
        const int m0 = col0 & (CL_NTRK - 1);
        const int m1 = col1 & (CL_NTRK - 1);
#pragma unroll
        for (int j = 0; j < 8; ++j) {
            const float e0 = __builtin_amdgcn_exp2f(c0[j] * CL_EXP2_SCALE);
            const float e1 = __builtin_amdgcn_exp2f(c1[j] * CL_EXP2_SCALE);
            tot[j] += e0 + e1;
            pos[j] += (m0 == trk[j]) ? e0 : 0.0f;
            pos[j] += (m1 == trk[j]) ? e1 : 0.0f;
        }
    }

#pragma unroll
    for (int j = 0; j < 8; ++j) {
        float t = tot[j], p = pos[j];
#pragma unroll
        for (int m = 1; m < 16; m <<= 1) {
            t += __shfl_xor(t, m, 32);
            p += __shfl_xor(p, m, 32);
        }
        if (r == j) {
            const int row = rowBase + 8 * h + j;
            ws[WS_XY_TOT + row * CHUNKS + colChunk] = t;
            ws[WS_XY_POS + row * CHUNKS + colChunk] = p;
        }
    }
}

/* ---------------- x @ x^T term ---------------- */
__global__ __launch_bounds__(256)
void cl_xx_kernel(const __bf16* __restrict__ xhi, const __bf16* __restrict__ xlo,
                  const int* __restrict__ trkIdx, float* __restrict__ ws) {
    const int lane = threadIdx.x & 31;
    const int wave = blockIdx.x * (blockDim.x >> 5) + (threadIdx.x >> 5);
    const int r = lane & 15;
    const int h = lane >> 4;
    const int rowTile  = wave >> 3;
    const int colChunk = wave & 7;        /* 8 chunks of 64 col tiles */
    const int rowBase  = rowTile * 16;

    v16bf Ahi[4], Alo[4];
    const __bf16* arowh = xhi + (size_t)(rowBase + r) * CL_D;
    const __bf16* arowl = xlo + (size_t)(rowBase + r) * CL_D;
#pragma unroll
    for (int t = 0; t < 4; ++t) {
        Ahi[t] = load_frag_A_bf(arowh, t, h);
        Alo[t] = load_frag_A_bf(arowl, t, h);
    }

    int trk[8];
#pragma unroll
    for (int j = 0; j < 8; ++j) trk[j] = trkIdx[rowBase + 8 * h + j];

    float tot[8], same[8], dia[8];
#pragma unroll
    for (int j = 0; j < 8; ++j) { tot[j] = 0.0f; same[j] = 0.0f; dia[j] = 0.0f; }

    const int ctBeg = colChunk * XX_TPC;
    const int ctEnd = ctBeg + XX_TPC;
    for (int ct = ctBeg; ct < ctEnd; ct += 2) {
        const int col0 = ct * 16 + r;
        const int col1 = col0 + 16;
        const __bf16* b0h = xhi + (size_t)col0 * CL_D;
        const __bf16* b0l = xlo + (size_t)col0 * CL_D;
        const __bf16* b1h = xhi + (size_t)col1 * CL_D;
        const __bf16* b1l = xlo + (size_t)col1 * CL_D;
        if (ct + 2 < ctEnd) {
            __builtin_prefetch(b0h + 32 * CL_D, 0, 1);
            __builtin_prefetch(b0l + 32 * CL_D, 0, 1);
        }
        const int t0 = trkIdx[col0];
        const int t1 = trkIdx[col1];
        const v8f c0 = tile_mm(b0h, b0l, Ahi, Alo, h);
        const v8f c1 = tile_mm(b1h, b1l, Ahi, Alo, h);
#pragma unroll
        for (int j = 0; j < 8; ++j) {
            const int row = rowBase + 8 * h + j;
            const float e0 = __builtin_amdgcn_exp2f(c0[j] * CL_EXP2_SCALE);
            const float e1 = __builtin_amdgcn_exp2f(c1[j] * CL_EXP2_SCALE);
            tot[j]  += e0 + e1;
            same[j] += (t0 == trk[j]) ? e0 : 0.0f;
            same[j] += (t1 == trk[j]) ? e1 : 0.0f;
            dia[j]  += (col0 == row) ? e0 : 0.0f;
            dia[j]  += (col1 == row) ? e1 : 0.0f;
        }
    }

#pragma unroll
    for (int j = 0; j < 8; ++j) {
        float t = tot[j], s = same[j], d = dia[j];
#pragma unroll
        for (int m = 1; m < 16; m <<= 1) {
            t += __shfl_xor(t, m, 32);
            s += __shfl_xor(s, m, 32);
            d += __shfl_xor(d, m, 32);
        }
        if (r == j) {
            const int row = rowBase + 8 * h + j;
            ws[WS_XX_TOT  + row * CHUNKS + colChunk] = t;
            ws[WS_XX_SAME + row * CHUNKS + colChunk] = s;
            ws[WS_XX_DIAG + row * CHUNKS + colChunk] = d;
        }
    }
}

/* ---------------- per-track loss + mean ---------------- */
__global__ __launch_bounds__(512)
void cl_finish_kernel(const float* __restrict__ ws, const int* __restrict__ trkIdx,
                      float* __restrict__ out) {
    __shared__ float red[512];
    const int i = threadIdx.x;  /* track id, 512 threads */

    float Pxy = 0.0f, Axy = 0.0f, Axx = 0.0f, G = 0.0f, Ds = 0.0f;
#pragma unroll 1
    for (int t = 0; t < CL_M / CL_NTRK; ++t) {
        const int row = i + CL_NTRK * t;
        if (trkIdx[row] == i) {
#pragma unroll
            for (int ch = 0; ch < CHUNKS; ++ch) {
                Axy += ws[WS_XY_TOT  + row * CHUNKS + ch];
                Pxy += ws[WS_XY_POS  + row * CHUNKS + ch];
                Axx += ws[WS_XX_TOT  + row * CHUNKS + ch];
                G   += ws[WS_XX_SAME + row * CHUNKS + ch];
                Ds  += ws[WS_XX_DIAG + row * CHUNKS + ch];
            }
        }
    }
    const float num = Pxy + 0.5f * (G - Ds);
    const float den = (Axy - Pxy) + (Axx - G);
    red[i] = logf((den + num) / num);   /* == -log(num/(den+num)) */
    __syncthreads();
#pragma unroll
    for (int s = 256; s > 0; s >>= 1) {
        if (i < s) red[i] += red[i + s];
        __syncthreads();
    }
    if (i == 0)
        out[0] = red[0] * (1.0f / (float)CL_NTRK) * (1.0f / (float)CL_Q);
}

extern "C" void kernel_launch(void* const* d_in, const int* in_sizes, int n_in,
                              void* d_out, int out_size, void* d_ws, size_t ws_size,
                              hipStream_t stream) {
    (void)in_sizes; (void)n_in; (void)out_size; (void)ws_size;
    const float* x   = (const float*)d_in[0];
    const int*   trk = (const int*)d_in[1];
    const float* y   = (const float*)d_in[2];   /* [512,8,128] -> flat [4096,128] */
    float*       ws  = (float*)d_ws;
    float*       out = (float*)d_out;

    char* wsb = (char*)d_ws;
    __bf16* xhi = (__bf16*)(wsb + WS_XHI_OFF);
    __bf16* xlo = (__bf16*)(wsb + WS_XLO_OFF);
    __bf16* yhi = (__bf16*)(wsb + WS_YHI_OFF);
    __bf16* ylo = (__bf16*)(wsb + WS_YLO_OFF);

    /* one-time precision split: fp32 -> bf16 hi + bf16 residual */
    const int n4x = CL_M * CL_D / 4;     /* 262144 */
    const int n4y = CL_NQ * CL_D / 4;    /* 131072 */
    cl_convert_kernel<<<dim3(n4x / 256), dim3(256), 0, stream>>>(
        x, (unsigned long long*)xhi, (unsigned long long*)xlo, n4x);
    cl_convert_kernel<<<dim3(n4y / 256), dim3(256), 0, stream>>>(
        y, (unsigned long long*)yhi, (unsigned long long*)ylo, n4y);

    /* 512 row-tiles x 8 column-chunks = 4096 waves = 512 blocks of 8 waves */
    cl_xy_kernel<<<dim3(ROW_TILES), dim3(256), 0, stream>>>(xhi, xlo, yhi, ylo, trk, ws);
    cl_xx_kernel<<<dim3(ROW_TILES), dim3(256), 0, stream>>>(xhi, xlo, trk, ws);
    cl_finish_kernel<<<dim3(1), dim3(512), 0, stream>>>(ws, trk, out);
}
